// CausalSelfAttention_83107617177910
// MI455X (gfx1250) — matrix operand order repaired and checked by reading, not compiled, not run on hardware
//
#include <hip/hip_runtime.h>
#include <cstdint>
#include <cstddef>

// ---------------------------------------------------------------------------
// CDNA5 (gfx1250) bf16 WMMA causal self-attention.
// D = A(16x32 bf16) x B(32x16 bf16) + C(16x16 f32), wave32.
// ---------------------------------------------------------------------------

typedef __attribute__((ext_vector_type(16))) __bf16 v16bf;
typedef __attribute__((ext_vector_type(8)))  float  v8f;

#define WMMA_BF16(a, b, c) \
    __builtin_amdgcn_wmma_f32_16x16x32_bf16(false, (a), false, (b), (short)0, (c), false, false)

// A-matrix fragment (16x32 bf16), source stored row-major [m][k], k contiguous.
// ISA layout: lane<16 row m=lane, K = e<8 ? e : e+8 ; lane>=16 same row, K += 8.
__device__ __forceinline__ v16bf load_afrag(const unsigned int* p, int rowStrideU,
                                            int m, int kU, int lane) {
    const unsigned int* row = p + (size_t)(m + (lane & 15)) * rowStrideU + kU;
    const int ko = (lane & 16) ? 4 : 0;   // uint units (8 bf16 = 4 uints)
    union { unsigned int u[8]; v16bf v; } t;
#pragma unroll
    for (int j = 0; j < 4; ++j) { t.u[j] = row[ko + j]; t.u[4 + j] = row[ko + 8 + j]; }
    return t.v;
}

// B-matrix fragment (32x16 bf16), source stored [n][k], k contiguous.
// ISA layout: lane<16 col n=lane, K = e<8 ? e : e+8 ; lane>=16 same col, K += 8.
__device__ __forceinline__ v16bf load_bfrag(const unsigned int* p, int rowStrideU,
                                            int n, int kU, int lane) {
    const unsigned int* row = p + (size_t)(n + (lane & 15)) * rowStrideU + kU;
    const int ko = (lane & 16) ? 4 : 0;
    union { unsigned int u[8]; v16bf v; } t;
#pragma unroll
    for (int j = 0; j < 4; ++j) { t.u[j] = row[ko + j]; t.u[4 + j] = row[ko + 8 + j]; }
    return t.v;
}

// ---------------------------------------------------------------------------
// fp32 -> bf16 convert (plain)
// ---------------------------------------------------------------------------
__global__ void f32_to_bf16_kernel(const float* __restrict__ in, __bf16* __restrict__ out, int n) {
    int i = blockIdx.x * 256 + threadIdx.x;
    if (i < n) out[i] = (__bf16)in[i];
}

// ---------------------------------------------------------------------------
// fp32 [K][N] -> bf16 [N][K] transpose-convert, 32x32 LDS tiles, 256 threads.
// Weights are transposed once so GEMM B-tiles are k-contiguous copies.
// ---------------------------------------------------------------------------
__global__ __launch_bounds__(256)
void transpose_f32_to_bf16_kernel(const float* __restrict__ in, __bf16* __restrict__ out,
                                  int K, int N) {
    __shared__ __bf16 tile[32][33];   // +1 pad to dodge bank conflicts
    const int kt = blockIdx.y * 32, nt = blockIdx.x * 32;
    const int tx = threadIdx.x & 31, ty = threadIdx.x >> 5;  // ty in 0..7
#pragma unroll
    for (int i = 0; i < 4; ++i) {
        int k = kt + ty + i * 8;
        tile[ty + i * 8][tx] = (__bf16)in[(size_t)k * N + nt + tx];
    }
    __syncthreads();
#pragma unroll
    for (int i = 0; i < 4; ++i) {
        int n = nt + ty + i * 8;
        out[(size_t)n * K + kt + tx] = tile[tx][ty + i * 8];
    }
}

// ---------------------------------------------------------------------------
// Tiled bf16 GEMM: C[M,N] = A[M,K] * Bt[N,K]^T, fp32 accumulate.
// 128x128 tile, K-step 32, 256 threads = 8 waves, each wave 32(M) x 64(N).
// Both LDS tiles are straight k-contiguous copies; next tile prefetched.
// ---------------------------------------------------------------------------
template <typename OutT>
__global__ __launch_bounds__(256)
void gemm_bf16_wmma(const __bf16* __restrict__ A, const __bf16* __restrict__ Bt,
                    OutT* __restrict__ C, int M, int N, int K) {
    __shared__ __bf16 sA[128 * 32];   // [m][k]
    __shared__ __bf16 sB[128 * 32];   // [n][k]

    const int tid  = threadIdx.x;
    const int lane = tid & 31;
    const int w    = tid >> 5;
    const int waveM = (w & 3) * 32;
    const int waveN = (w >> 2) * 64;
    const int mBlock = blockIdx.y * 128;
    const int nBlock = blockIdx.x * 128;

    v8f acc[2][4];
#pragma unroll
    for (int mi = 0; mi < 2; ++mi)
#pragma unroll
        for (int ni = 0; ni < 4; ++ni) acc[mi][ni] = {};

    const unsigned int* Au  = (const unsigned int*)A;
    const unsigned int* Btu = (const unsigned int*)Bt;
    const int Ku = K >> 1;
    unsigned int* sAu = (unsigned int*)sA;
    unsigned int* sBu = (unsigned int*)sB;

    for (int kt = 0; kt < K; kt += 32) {
        // A tile: 128 m-rows x 16 uints ; B tile: 128 n-rows x 16 uints
#pragma unroll
        for (int it = 0; it < 8; ++it) {
            int u = it * 256 + tid;
            int row = u >> 4, cu = u & 15;
            sAu[u] = Au[(size_t)(mBlock + row) * Ku + (kt >> 1) + cu];
            sBu[u] = Btu[(size_t)(nBlock + row) * Ku + (kt >> 1) + cu];
        }
        if (kt + 32 < K) {  // prefetch next K-tile (global_prefetch_b8)
            int row = tid >> 4, cu = tid & 15;
            __builtin_prefetch(&Au[(size_t)(mBlock + row) * Ku + ((kt + 32) >> 1) + cu], 0, 3);
            __builtin_prefetch(&Btu[(size_t)(nBlock + row) * Ku + ((kt + 32) >> 1) + cu], 0, 3);
        }
        __syncthreads();

        v16bf af[2], bf[4];
#pragma unroll
        for (int mi = 0; mi < 2; ++mi)
            af[mi] = load_afrag(sAu, 16, waveM + mi * 16, 0, lane);
#pragma unroll
        for (int ni = 0; ni < 4; ++ni)
            bf[ni] = load_bfrag(sBu, 16, waveN + ni * 16, 0, lane);
#pragma unroll
        for (int mi = 0; mi < 2; ++mi)
#pragma unroll
            for (int ni = 0; ni < 4; ++ni)
                acc[mi][ni] = WMMA_BF16(af[mi], bf[ni], acc[mi][ni]);
        __syncthreads();
    }

    const int rAdd = (lane & 16) ? 8 : 0;
    const int colL = lane & 15;
#pragma unroll
    for (int mi = 0; mi < 2; ++mi)
#pragma unroll
        for (int ni = 0; ni < 4; ++ni) {
            int m0  = mBlock + waveM + mi * 16 + rAdd;
            int col = nBlock + waveN + ni * 16 + colL;
#pragma unroll
            for (int r = 0; r < 8; ++r)
                C[(size_t)(m0 + r) * N + col] = (OutT)acc[mi][ni][r];
        }
}

// ---------------------------------------------------------------------------
// RoPE + split: qkv_bf16 [B,T,3C] -> q,k bf16 [B,H,T,D]; v bf16 [B,H,D,T]
// (V pre-transposed so the attention kernel's [d][key] LDS tile is a copy).
// 1/sqrt(D) folded into q.
// ---------------------------------------------------------------------------
__global__ __launch_bounds__(256)
void rope_split_kernel(const __bf16* __restrict__ qkv, __bf16* __restrict__ q,
                       __bf16* __restrict__ k, __bf16* __restrict__ v) {
    int idx = blockIdx.x * 256 + threadIdx.x;  // B*T*H*64 = 8388608
    int j = idx & 63;
    int h = (idx >> 6) & 15;
    int t = (idx >> 10) & 2047;
    int b = idx >> 21;

    const size_t ib = (size_t)(b * 2048 + t) * 6144;
    const int cq = h * 128 + j;
    float q1 = (float)qkv[ib + cq],        q2 = (float)qkv[ib + cq + 64];
    float k1 = (float)qkv[ib + 2048 + cq], k2 = (float)qkv[ib + 2048 + cq + 64];
    float v1 = (float)qkv[ib + 4096 + cq], v2 = (float)qkv[ib + 4096 + cq + 64];

    float ang = (float)t * __expf(-0.14391157f * (float)j);  // ln(1e4)/64
    float sn, cs; __sincosf(ang, &sn, &cs);
    const float scale = 0.08838834764831845f;                 // 1/sqrt(128)

    const size_t ob = ((size_t)(b * 16 + h) * 2048 + t) * 128;
    q[ob + j]      = (__bf16)((q1 * cs - q2 * sn) * scale);
    q[ob + j + 64] = (__bf16)((q2 * cs + q1 * sn) * scale);
    k[ob + j]      = (__bf16)(k1 * cs - k2 * sn);
    k[ob + j + 64] = (__bf16)(k2 * cs + k1 * sn);

    const size_t vbase = ((size_t)(b * 16 + h) * 128) * 2048 + t;  // [B,H,D,T]
    v[vbase + (size_t)j * 2048]        = (__bf16)v1;
    v[vbase + (size_t)(j + 64) * 2048] = (__bf16)v2;
}

// ---------------------------------------------------------------------------
// Flash attention: per (b,h, 64-row q tile). 128 threads = 4 waves, each wave
// owns 16 q rows. Online softmax, causal mask on diagonal tile only.
// K: [B,H,T,D] ; V: [B,H,D,T] (pre-transposed) ; Y: [B,T,C]
// ---------------------------------------------------------------------------
__global__ __launch_bounds__(128)
void flash_attn_kernel(const __bf16* __restrict__ Q, const __bf16* __restrict__ Kh,
                       const __bf16* __restrict__ Vt, __bf16* __restrict__ Y) {
    __shared__ __bf16 sK[64 * 128];    // [key][d]
    __shared__ __bf16 sVt[128 * 64];   // [d][key]
    __shared__ __bf16 sP[4][16 * 64];  // per-wave P tile

    const int tid  = threadIdx.x;
    const int lane = tid & 31;
    const int w    = tid >> 5;
    const int qt   = blockIdx.x;  // 0..31
    const int bh   = blockIdx.y;  // 0..63
    const size_t headBase = (size_t)bh * 2048 * 128;
    const int qRow0 = qt * 64 + w * 16;

    // Q A-fragments: 4 chunks of K=32 over D=128, straight from global [t][d].
    const unsigned int* Qu = (const unsigned int*)(Q + headBase);
    v16bf qf[4];
#pragma unroll
    for (int c = 0; c < 4; ++c) qf[c] = load_afrag(Qu, 64, qRow0, c * 16, lane);

    v8f o[8];
#pragma unroll
    for (int df = 0; df < 8; ++df) o[df] = {};
    float mrow[8], lrow[8];
#pragma unroll
    for (int r = 0; r < 8; ++r) { mrow[r] = -1e30f; lrow[r] = 0.f; }

    const int rAdd = (lane & 16) ? 8 : 0;
    const int myQ  = qt * 64 + w * 16 + rAdd;

    unsigned int* sKu  = (unsigned int*)sK;
    unsigned int* sVtu = (unsigned int*)sVt;
    const unsigned int* KuG  = (const unsigned int*)(Kh + headBase);
    const unsigned int* VtuG = (const unsigned int*)(Vt + headBase);

    for (int j = 0; j <= qt; ++j) {
        const int kv0 = j * 64;
        // K tile: rows t contiguous -> straight copy of 4096 uints.
#pragma unroll
        for (int it = 0; it < 32; ++it) {
            int u = it * 128 + tid;
            sKu[u] = KuG[(size_t)kv0 * 64 + u];
        }
        // V tile: global rows d (stride T), 32 uints of keys each -> copy.
#pragma unroll
        for (int it = 0; it < 32; ++it) {
            int u = it * 128 + tid;          // d = u>>5, cu = u&31
            int d = u >> 5, cu = u & 31;
            sVtu[u] = VtuG[(size_t)d * 1024 + (kv0 >> 1) + cu];
        }
        __syncthreads();

        // S = Q * K^T (scale pre-folded into Q)
        v8f s[4];
#pragma unroll
        for (int nk = 0; nk < 4; ++nk) s[nk] = {};
#pragma unroll
        for (int nk = 0; nk < 4; ++nk)
#pragma unroll
            for (int c = 0; c < 4; ++c) {
                v16bf bf = load_bfrag(sKu, 64, nk * 16, c * 16, lane);
                s[nk] = WMMA_BF16(qf[c], bf, s[nk]);
            }

        if (j == qt) {  // causal mask on diagonal tile
#pragma unroll
            for (int nk = 0; nk < 4; ++nk) {
                int kidx = kv0 + nk * 16 + (lane & 15);
#pragma unroll
                for (int r = 0; r < 8; ++r)
                    if (kidx > myQ + r) s[nk][r] = -1e30f;
            }
        }

        // online softmax (rows of a fragment live in one 16-lane half)
#pragma unroll
        for (int r = 0; r < 8; ++r) {
            float mt = fmaxf(fmaxf(s[0][r], s[1][r]), fmaxf(s[2][r], s[3][r]));
            mt = fmaxf(mt, __shfl_xor(mt, 1));
            mt = fmaxf(mt, __shfl_xor(mt, 2));
            mt = fmaxf(mt, __shfl_xor(mt, 4));
            mt = fmaxf(mt, __shfl_xor(mt, 8));
            float mn = fmaxf(mrow[r], mt);
            float alpha = __expf(mrow[r] - mn);
            mrow[r] = mn;
            float rs = 0.f;
#pragma unroll
            for (int nk = 0; nk < 4; ++nk) {
                float p = __expf(s[nk][r] - mn);
                s[nk][r] = p;
                rs += p;
            }
            rs += __shfl_xor(rs, 1); rs += __shfl_xor(rs, 2);
            rs += __shfl_xor(rs, 4); rs += __shfl_xor(rs, 8);
            lrow[r] = lrow[r] * alpha + rs;
#pragma unroll
            for (int df = 0; df < 8; ++df) o[df][r] *= alpha;
        }

        // P: C-layout regs -> per-wave LDS -> A-layout fragments
        __bf16* pw = sP[w];
        {
            const int colbase = lane & 15;
#pragma unroll
            for (int nk = 0; nk < 4; ++nk)
#pragma unroll
                for (int r = 0; r < 8; ++r)
                    pw[(r + rAdd) * 64 + nk * 16 + colbase] = (__bf16)s[nk][r];
        }
        asm volatile("s_wait_dscnt 0" ::: "memory");

        v16bf pa[2];
#pragma unroll
        for (int c = 0; c < 2; ++c)
            pa[c] = load_afrag((const unsigned int*)pw, 32, 0, c * 16, lane);

        // O += P * V
#pragma unroll
        for (int df = 0; df < 8; ++df)
#pragma unroll
            for (int c = 0; c < 2; ++c) {
                v16bf bf = load_bfrag(sVtu, 32, df * 16, c * 16, lane);
                o[df] = WMMA_BF16(pa[c], bf, o[df]);
            }
        __syncthreads();
    }

    // normalize and write y [B,T,C] bf16
    const int b = bh >> 4, h = bh & 15;
    __bf16* yout = Y + (size_t)b * 2048 * 2048 + h * 128;
    const int colL = lane & 15;
#pragma unroll
    for (int r = 0; r < 8; ++r) {
        float inv = 1.f / lrow[r];
        int t = qt * 64 + w * 16 + r + rAdd;
#pragma unroll
        for (int df = 0; df < 8; ++df)
            yout[(size_t)t * 2048 + df * 16 + colL] = (__bf16)(o[df][r] * inv);
    }
}

// ---------------------------------------------------------------------------
// Launch
// ---------------------------------------------------------------------------
extern "C" void kernel_launch(void* const* d_in, const int* in_sizes, int n_in,
                              void* d_out, int out_size, void* d_ws, size_t ws_size,
                              hipStream_t stream) {
    (void)in_sizes; (void)n_in; (void)out_size; (void)ws_size;
    const float* x      = (const float*)d_in[0];  // [4,2048,2048]
    const float* w_attn = (const float*)d_in[1];  // [2048,6144]
    const float* w_proj = (const float*)d_in[2];  // [2048,2048]
    float* out = (float*)d_out;                   // [4,2048,2048]

    const int B = 4, T = 2048, C = 2048;
    const int M = B * T;            // 8192
    const int N_QKV = 3 * C;        // 6144
    const size_t nX  = (size_t)M * C;
    const size_t nWA = (size_t)C * N_QKV;
    const size_t nWP = (size_t)C * C;
    const size_t nQKV = (size_t)M * N_QKV;

    char* ws = (char*)d_ws;
    size_t off = 0;
    auto take = [&](size_t bytes) { char* p = ws + off; off += (bytes + 255) & ~(size_t)255; return p; };

    __bf16* xb   = (__bf16*)take(nX * 2);      //  32 MB  x bf16 [M][K]
    __bf16* wabT = (__bf16*)take(nWA * 2);     //  24 MB  w_attn bf16 [N][K]
    __bf16* wpbT = (__bf16*)take(nWP * 2);     //   8 MB  w_proj bf16 [N][K]
    char*   qkvp = take(nQKV * 2);             // 100 MB  (reused for y after rope)
    __bf16* qkvb = (__bf16*)qkvp;
    __bf16* qb   = (__bf16*)take(nX * 2);      //  32 MB  [B,H,T,D]
    __bf16* kb   = (__bf16*)take(nX * 2);      //  32 MB  [B,H,T,D]
    __bf16* vb   = (__bf16*)take(nX * 2);      //  32 MB  [B,H,D,T]
    __bf16* yb   = (__bf16*)qkvp;              // alias dead qkv buffer

    f32_to_bf16_kernel<<<(int)(nX / 256), 256, 0, stream>>>(x, xb, (int)nX);
    transpose_f32_to_bf16_kernel<<<dim3(N_QKV / 32, C / 32), 256, 0, stream>>>(
        w_attn, wabT, C, N_QKV);
    transpose_f32_to_bf16_kernel<<<dim3(C / 32, C / 32), 256, 0, stream>>>(
        w_proj, wpbT, C, C);

    // qkv = x @ w_attn  (bf16 out)
    gemm_bf16_wmma<__bf16><<<dim3(N_QKV / 128, M / 128), 256, 0, stream>>>(
        xb, wabT, qkvb, M, N_QKV, C);

    // RoPE + split to head-major layouts
    rope_split_kernel<<<(int)(nX / 2 / 256), 256, 0, stream>>>(qkvb, qb, kb, vb);

    // flash attention -> y [B,T,C] bf16
    flash_attn_kernel<<<dim3(T / 64, B * 16), 128, 0, stream>>>(qb, kb, vb, yb);

    // out = y @ w_proj (f32 out)
    gemm_bf16_wmma<float><<<dim3(C / 128, M / 128), 256, 0, stream>>>(
        yb, wpbT, out, M, C, C);
}
